// SFT_adaptive_37314675867933
// MI455X (gfx1250) — matrix operand order repaired and checked by reading, not compiled, not run on hardware
//
#include <hip/hip_runtime.h>
#include <hip/hip_bf16.h>
#include <stdint.h>

// ---------------------------------------------------------------------------
// SFT_adaptive forward for MI455X (gfx1250, wave32, WMMA).
// All large einsums run as bf16 WMMA GEMMs (v_wmma_f32_16x16x32_bf16),
// wide-K reductions use split-K + global f32 atomics, small glue in f32.
// LDS layouts are arranged so every WMMA fragment is two contiguous
// 16B ds_load_b128 reads (no VGPR shuffle). Async global->LDS DMA
// (ASYNCcnt) is used for the NT GEMM staging when the toolchain has it.
// ---------------------------------------------------------------------------

typedef __bf16 vbf16 __attribute__((ext_vector_type(16)));
typedef float  vf8   __attribute__((ext_vector_type(8)));
typedef int    v4i   __attribute__((vector_size(16)));   // matches builtin param

#define ASG __attribute__((address_space(1)))
#define ASL __attribute__((address_space(3)))

#if defined(__AMDGCN__) && __has_builtin(__builtin_amdgcn_global_load_async_to_lds_b128) && __has_builtin(__builtin_amdgcn_s_wait_asynccnt)
#define USE_ASYNC_LDS 1
#else
#define USE_ASYNC_LDS 0
#endif

__device__ __forceinline__ unsigned short f2bf(float f) {
  unsigned u = __builtin_bit_cast(unsigned, f);
  unsigned r = u + 0x7FFFu + ((u >> 16) & 1u);   // round-to-nearest-even
  return (unsigned short)(r >> 16);
}

// ======================= WMMA GEMM (NN): C = A(f32,MxK) * B(bf16,KxN) ======
// Block: 256 threads (8 waves). Tile: 64(M) x 128(N). K staged in LDS chunks
// of 64. B staged as [n][kpair] (pad 36) so each B fragment is 2x b128 reads.
__global__ __launch_bounds__(256) void k_gemm_nn(
    const float* __restrict__ A, const unsigned short* __restrict__ B,
    const float* __restrict__ bias, void* __restrict__ Cout,
    int M, int K, int N, int out_is_bf16)
{
  __shared__ unsigned short As[64 * 64];    // [m][k] bf16
  __shared__ unsigned int   Bs[128 * 36];   // [n][kpair] packed {k,k+1}, pad 36

  const int tid  = threadIdx.x;
  const int wave = tid >> 5;
  const int lane = tid & 31;
  const int hi   = lane >> 4;
  const int ln   = lane & 15;

  const int  m0 = blockIdx.y * 64;
  const long n0 = (long)blockIdx.x * 128;

  const int mrow0 = (wave & 3) * 16;
  const int nwav0 = (wave >> 2) * 64;

  vf8 acc[4];
#pragma unroll
  for (int t = 0; t < 4; ++t)
#pragma unroll
    for (int j = 0; j < 8; ++j) acc[t][j] = 0.f;

  for (int kc = 0; kc < K; kc += 64) {
    // ---- stage A chunk (f32 -> bf16, row-major [m][k]) ----
    for (int i = tid; i < 4096; i += 256) {
      int r = i >> 6, c = i & 63;
      As[i] = f2bf(A[(size_t)(m0 + r) * K + (kc + c)]);
    }
    // ---- stage B chunk: pack K-pairs, transpose to [n][kpair] ----
    for (int i = tid; i < 1024; i += 256) {
      int p  = i >> 5;            // k-pair 0..31
      int ng = (i & 31) * 4;      // n group of 4
      size_t base = (size_t)(kc + p * 2) * N + (size_t)(n0 + ng);
      ushort4 r0 = *(const ushort4*)(B + base);
      ushort4 r1 = *(const ushort4*)(B + base + N);
      Bs[(ng + 0) * 36 + p] = (unsigned)r0.x | ((unsigned)r1.x << 16);
      Bs[(ng + 1) * 36 + p] = (unsigned)r0.y | ((unsigned)r1.y << 16);
      Bs[(ng + 2) * 36 + p] = (unsigned)r0.z | ((unsigned)r1.z << 16);
      Bs[(ng + 3) * 36 + p] = (unsigned)r0.w | ((unsigned)r1.w << 16);
    }
    if (kc + 64 < K) {            // gfx1250 global_prefetch_b8 for next chunk
      __builtin_prefetch(B + (size_t)(kc + 64) * N + n0, 0, 1);
      __builtin_prefetch(A + (size_t)m0 * K + (kc + 64), 0, 1);
    }
    __syncthreads();

    // ---- two WMMA K-steps of 32 ----
#pragma unroll
    for (int ks = 0; ks < 64; ks += 32) {
      union { unsigned u[8]; uint4 q[2]; vbf16 v; } af;
      // ISA 16-bit A 16x32 layout -> two contiguous 16B reads per lane
      const unsigned short* ap = &As[(mrow0 + ln) * 64 + ks + hi * 8];
      af.q[0] = *(const uint4*)ap;
      af.q[1] = *(const uint4*)(ap + 16);
#pragma unroll
      for (int t = 0; t < 4; ++t) {
        union { unsigned u[8]; uint4 q[2]; vbf16 v; } bfg;
        int ncol  = nwav0 + t * 16 + ln;
        int pbase = ncol * 36 + (ks >> 1) + hi * 4;  // B 32x16: K = hi*8+2v (v<4), 16+hi*8+2(v-4)
        bfg.q[0] = *(const uint4*)&Bs[pbase];
        bfg.q[1] = *(const uint4*)&Bs[pbase + 8];
        acc[t] = __builtin_amdgcn_wmma_f32_16x16x32_bf16(
            false, af.v, false, bfg.v, (short)0, acc[t], false, false);
      }
    }
    __syncthreads();
  }

  // ---- epilogue: C layout VGPR r -> M = r + hi*8, N = lane&15 ----
#pragma unroll
  for (int t = 0; t < 4; ++t) {
#pragma unroll
    for (int r = 0; r < 8; ++r) {
      int  m = m0 + mrow0 + r + hi * 8;
      long n = n0 + nwav0 + t * 16 + ln;
      float v = acc[t][r];
      if (bias) v += bias[m];
      if (out_is_bf16) ((unsigned short*)Cout)[(size_t)m * N + n] = f2bf(v);
      else             ((float*)Cout)[(size_t)m * N + n] = v;
    }
  }
}

// ============ WMMA GEMM (NT, split-K): C += A(bf16,MxK) * B(bf16,NxK)^T ====
// Block tile 64x64, grid.z = K-chunks, f32 atomic-add epilogue (C pre-zeroed).
// Staging is a layout-identical row copy -> async global->LDS DMA when
// available (ASYNCcnt + s_wait_asynccnt), plain loads otherwise.
__global__ __launch_bounds__(256) void k_gemm_nt(
    const unsigned short* __restrict__ A, const unsigned short* __restrict__ B,
    float* __restrict__ C, int M, int N, long K, long kChunk)
{
  __shared__ unsigned short As[64 * 64];   // [m][k]
  __shared__ unsigned short Bsm[64 * 64];  // [n][k]  (row of B == column of B^T)

  const int tid  = threadIdx.x;
  const int wave = tid >> 5;
  const int lane = tid & 31;
  const int hi   = lane >> 4;
  const int ln   = lane & 15;

  const int m0 = blockIdx.y * 64;
  const int n0 = blockIdx.x * 64;
  long kstart = (long)blockIdx.z * kChunk;
  long kend   = kstart + kChunk; if (kend > K) kend = K;

  const int mrow0 = (wave & 3) * 16;
  const int nw    = (wave >> 2);          // 2 n-tiles per wave

  vf8 acc[2];
#pragma unroll
  for (int t = 0; t < 2; ++t)
#pragma unroll
    for (int j = 0; j < 8; ++j) acc[t][j] = 0.f;

  for (long kc = kstart; kc < kend; kc += 64) {
#if USE_ASYNC_LDS
    for (int i = tid; i < 512; i += 256) {
      int r = i >> 3, c8 = (i & 7) * 8;   // 16-byte granules
      __builtin_amdgcn_global_load_async_to_lds_b128(
          (ASG v4i*)(A + (size_t)(m0 + r) * K + kc + c8),
          (ASL v4i*)&As[r * 64 + c8], 0, 0);
      __builtin_amdgcn_global_load_async_to_lds_b128(
          (ASG v4i*)(B + (size_t)(n0 + r) * K + kc + c8),
          (ASL v4i*)&Bsm[r * 64 + c8], 0, 0);
    }
    __builtin_amdgcn_s_wait_asynccnt(0);
#else
    for (int i = tid; i < 1024; i += 256) {
      int r = i >> 4, c4 = (i & 15) * 4;
      *(ushort4*)&As [r * 64 + c4] = *(const ushort4*)(A + (size_t)(m0 + r) * K + kc + c4);
      *(ushort4*)&Bsm[r * 64 + c4] = *(const ushort4*)(B + (size_t)(n0 + r) * K + kc + c4);
    }
#endif
    if (kc + 64 < kend) {
      __builtin_prefetch(A + (size_t)m0 * K + kc + 64, 0, 1);
      __builtin_prefetch(B + (size_t)n0 * K + kc + 64, 0, 1);
    }
    __syncthreads();

#pragma unroll
    for (int ks = 0; ks < 64; ks += 32) {
      union { unsigned u[8]; uint4 q[2]; vbf16 v; } af;
      const unsigned short* ap = &As[(mrow0 + ln) * 64 + ks + hi * 8];
      af.q[0] = *(const uint4*)ap;
      af.q[1] = *(const uint4*)(ap + 16);
#pragma unroll
      for (int t = 0; t < 2; ++t) {
        union { unsigned u[8]; uint4 q[2]; vbf16 v; } bfg;
        int ncol = (nw * 2 + t) * 16 + ln;
        const unsigned short* bp = &Bsm[ncol * 64 + ks + hi * 8];
        bfg.q[0] = *(const uint4*)bp;        // consecutive K pairs from B row
        bfg.q[1] = *(const uint4*)(bp + 16);
        acc[t] = __builtin_amdgcn_wmma_f32_16x16x32_bf16(
            false, af.v, false, bfg.v, (short)0, acc[t], false, false);
      }
    }
    __syncthreads();
  }

#pragma unroll
  for (int t = 0; t < 2; ++t) {
#pragma unroll
    for (int r = 0; r < 8; ++r) {
      int m = m0 + mrow0 + r + hi * 8;
      int n = n0 + (nw * 2 + t) * 16 + ln;
      __hip_atomic_fetch_add(&C[(size_t)m * N + n], acc[t][r],
                             __ATOMIC_RELAXED, __HIP_MEMORY_SCOPE_AGENT);
    }
  }
}

// ============================== glue kernels ===============================
__global__ void k_zero(float* p, long n) {
  long i = (long)blockIdx.x * blockDim.x + threadIdx.x;
  for (; i < n; i += (long)gridDim.x * blockDim.x) p[i] = 0.f;
}

__global__ void k_cvt(const float* __restrict__ x, unsigned short* __restrict__ o, long n) {
  long i = (long)blockIdx.x * blockDim.x + threadIdx.x;
  if (i < n) o[i] = f2bf(x[i]);
}

// T[d][(c,h,w)] = x[(c,h,w)][d]   -> bf16 (64 x 262144)
__global__ void k_transp(const float* __restrict__ x, unsigned short* __restrict__ T) {
  long i = (long)blockIdx.x * blockDim.x + threadIdx.x;
  if (i >= 16777216L) return;
  long d = i >> 18, m = i & 262143;
  T[i] = f2bf(x[m * 64 + d]);
}

// depthwise 3x3x3 stride-2: x(64,64,64,64) -> Hs(64,32,32,32) f32 + bf16 copy
__global__ void k_dwconv(const float* __restrict__ x, const float* __restrict__ w,
                         const float* __restrict__ b, float* __restrict__ Hs,
                         unsigned short* __restrict__ Hsbf) {
  long i = (long)blockIdx.x * blockDim.x + threadIdx.x;
  if (i >= 2097152L) return;
  int dd = i & 31, ww = (i >> 5) & 31, hh = (i >> 10) & 31, c = (int)(i >> 15);
  float a = b[c];
  for (int kh = 0; kh < 3; ++kh) {
    int ih = hh * 2 - 1 + kh; if (ih < 0 || ih > 63) continue;
    for (int kw = 0; kw < 3; ++kw) {
      int iw = ww * 2 - 1 + kw; if (iw < 0 || iw > 63) continue;
      for (int kd = 0; kd < 3; ++kd) {
        int id = dd * 2 - 1 + kd; if (id < 0 || id > 63) continue;
        a += w[c * 27 + kh * 9 + kw * 3 + kd] *
             x[(((size_t)c * 64 + ih) * 64 + iw) * 64 + id];
      }
    }
  }
  Hs[i] = a; Hsbf[i] = f2bf(a);
}

// ps[(o,w),(c,d)] = sum_h phi_s_w[o,h]*Hs[c,h,w,d] + b[o]  (bf16 1024x2048)
__global__ void k_ps(const float* __restrict__ Hs, const float* __restrict__ w,
                     const float* __restrict__ b, unsigned short* __restrict__ ps) {
  long i = (long)blockIdx.x * blockDim.x + threadIdx.x;
  if (i >= 2097152L) return;
  int col = (int)(i & 2047), row = (int)(i >> 11);
  int o = row >> 5, ww = row & 31, c = col >> 5, dd = col & 31;
  float a = b[o];
  for (int h = 0; h < 32; ++h)
    a += w[o * 32 + h] * Hs[((size_t)c * 32 + h) * 1024 + ww * 32 + dd];
  ps[i] = f2bf(a);
}

// vv[(o,h),(c,d)] = sum_w v_w[o,w]*Hs[c,h,w,d] + vb[o]  (bf16 1024x2048)
__global__ void k_vv(const float* __restrict__ Hs, const float* __restrict__ w,
                     const float* __restrict__ b, unsigned short* __restrict__ vv) {
  long i = (long)blockIdx.x * blockDim.x + threadIdx.x;
  if (i >= 2097152L) return;
  int col = (int)(i & 2047), row = (int)(i >> 11);
  int o = row >> 5, hh = row & 31, c = col >> 5, dd = col & 31;
  float a = b[o];
  for (int wq = 0; wq < 32; ++wq)
    a += w[o * 32 + wq] * Hs[((size_t)c * 32 + hh) * 1024 + wq * 32 + dd];
  vv[i] = f2bf(a);
}

__global__ __launch_bounds__(256) void k_softmax(const float* __restrict__ Al,
                                                 unsigned short* __restrict__ Ao) {
  __shared__ float red[256];
  int r = blockIdx.x, tid = threadIdx.x;
  const float* row = Al + (size_t)r * 1024;
  float mx = -3.4e38f;
  for (int j = tid; j < 1024; j += 256) mx = fmaxf(mx, row[j]);
  red[tid] = mx; __syncthreads();
  for (int s = 128; s > 0; s >>= 1) { if (tid < s) red[tid] = fmaxf(red[tid], red[tid + s]); __syncthreads(); }
  mx = red[0]; __syncthreads();
  float e[4]; float sm = 0.f;
  for (int q = 0; q < 4; ++q) { e[q] = __expf(row[tid + q * 256] - mx); sm += e[q]; }
  red[tid] = sm; __syncthreads();
  for (int s = 128; s > 0; s >>= 1) { if (tid < s) red[tid] += red[tid + s]; __syncthreads(); }
  float inv = 1.f / red[0];
  for (int q = 0; q < 4; ++q) Ao[(size_t)r * 1024 + tid + q * 256] = f2bf(e[q] * inv);
}

// AVs_mat(2048x1024)[(c*32+d), (h*32+w)] -> rp bf16 [c][(h,w,d)] (64 x 32768)
__global__ void k_repack(const float* __restrict__ av, unsigned short* __restrict__ rp) {
  long i = (long)blockIdx.x * blockDim.x + threadIdx.x;
  if (i >= 2097152L) return;
  int dd = i & 31, ww = (i >> 5) & 31, hh = (i >> 10) & 31, c = (int)(i >> 15);
  rp[i] = f2bf(av[(size_t)(c * 32 + dd) * 1024 + hh * 32 + ww]);
}

// GCN: out = w2 @ relu(g @ w1^T + b1 + g)   (S = 64 or 128, f32, one block)
__global__ __launch_bounds__(256) void k_gcn(const float* __restrict__ g,
                                             const float* __restrict__ w1,
                                             const float* __restrict__ b1,
                                             const float* __restrict__ w2,
                                             float* __restrict__ out, int S) {
  extern __shared__ float hsh[];
  int tid = threadIdx.x;
  for (int i = tid; i < S * S; i += 256) {
    int s = i / S, o = i - s * S;
    float a = b1[o] + g[i];
    for (int n = 0; n < S; ++n) a += w1[o * S + n] * g[s * S + n];
    hsh[i] = a > 0.f ? a : 0.f;
  }
  __syncthreads();
  for (int i = tid; i < S * S; i += 256) {
    int o = i / S, n = i - o * S;
    float a = 0.f;
    for (int s = 0; s < S; ++s) a += w2[o * S + s] * hsh[s * S + n];
    out[i] = a;
  }
}

// per-row mean/var over 262144 cols: rows 0..63 -> yf, 64..127 -> yt
__global__ __launch_bounds__(256) void k_rowstats(const float* __restrict__ yf,
                                                  const float* __restrict__ yt,
                                                  float* __restrict__ stats) {
  __shared__ float s1[256], s2[256];
  int r = blockIdx.x, tid = threadIdx.x;
  const float* row = (r < 64) ? (yf + (size_t)r * 262144)
                              : (yt + (size_t)(r - 64) * 262144);
  float a = 0.f, b = 0.f;
  for (int j = tid; j < 262144; j += 256) { float v = row[j]; a += v; b += v * v; }
  s1[tid] = a; s2[tid] = b; __syncthreads();
  for (int s = 128; s > 0; s >>= 1) {
    if (tid < s) { s1[tid] += s1[tid + s]; s2[tid] += s2[tid + s]; }
    __syncthreads();
  }
  if (tid == 0) {
    float mean = s1[0] * (1.f / 262144.f);
    float var  = s2[0] * (1.f / 262144.f) - mean * mean;
    stats[r * 2] = mean; stats[r * 2 + 1] = var;
  }
}

// out = x + w1*up2(osp_half) + w2*BN(yf) + w3*BN(yt^T)
__global__ void k_final(const float* __restrict__ x, const float* __restrict__ yf,
                        const float* __restrict__ yt, const float* __restrict__ osp,
                        const float* __restrict__ stats,
                        const float* __restrict__ gF, const float* __restrict__ bF,
                        const float* __restrict__ gT, const float* __restrict__ bT,
                        const float* __restrict__ w1, const float* __restrict__ w2,
                        const float* __restrict__ w3, float* __restrict__ out) {
  long i = (long)blockIdx.x * blockDim.x + threadIdx.x;
  if (i >= 16777216L) return;
  int d = i & 63, w = (int)((i >> 6) & 63), h = (int)((i >> 12) & 63), c = (int)(i >> 18);
  long rem = i & 262143;
  float vf = yf[(size_t)c * 262144 + rem];
  float nf = gF[c] * (vf - stats[c * 2]) * rsqrtf(stats[c * 2 + 1] + 1e-5f) + bF[c];
  long ncol = (long)c * 4096 + h * 64 + w;
  float vt = yt[(size_t)d * 262144 + ncol];
  float ntv = gT[d] * (vt - stats[128 + d * 2]) * rsqrtf(stats[128 + d * 2 + 1] + 1e-5f) + bT[d];
  float ov = osp[(size_t)c * 32768 + (size_t)(h >> 1) * 1024 + (w >> 1) * 32 + (d >> 1)];
  out[i] = x[i] + w1[0] * ov + w2[0] * nf + w3[0] * ntv;
}

// ================================ host =====================================
extern "C" void kernel_launch(void* const* d_in, const int* in_sizes, int n_in,
                              void* d_out, int out_size, void* d_ws, size_t ws_size,
                              hipStream_t stream) {
  const float* x        = (const float*)d_in[0];
  const float* sproj_w  = (const float*)d_in[1];
  const float* sproj_b  = (const float*)d_in[2];
  const float* phi_s_w  = (const float*)d_in[3];
  const float* phi_s_b  = (const float*)d_in[4];
  const float* v_w      = (const float*)d_in[5];
  const float* v_b      = (const float*)d_in[6];
  const float* delta_w  = (const float*)d_in[7];
  const float* delta_b  = (const float*)d_in[8];
  const float* Ws_w     = (const float*)d_in[9];
  const float* Ws_b     = (const float*)d_in[10];
  const float* xi_w     = (const float*)d_in[11];
  const float* xi_b     = (const float*)d_in[12];
  const float* phi_f_w  = (const float*)d_in[13];
  const float* phi_f_b  = (const float*)d_in[14];
  const float* th_f_w   = (const float*)d_in[15];
  const float* th_f_b   = (const float*)d_in[16];
  const float* phi_t_w  = (const float*)d_in[17];
  const float* phi_t_b  = (const float*)d_in[18];
  const float* th_t_w   = (const float*)d_in[19];
  const float* th_t_b   = (const float*)d_in[20];
  const float* gcnf_w1  = (const float*)d_in[21];
  const float* gcnf_b1  = (const float*)d_in[22];
  const float* gcnf_w2  = (const float*)d_in[23];
  const float* gcnt_w1  = (const float*)d_in[24];
  const float* gcnt_b1  = (const float*)d_in[25];
  const float* gcnt_w2  = (const float*)d_in[26];
  const float* ext_f_w  = (const float*)d_in[27];
  const float* ext_f_b  = (const float*)d_in[28];
  const float* ext_f_g  = (const float*)d_in[29];
  const float* ext_f_be = (const float*)d_in[30];
  const float* ext_t_w  = (const float*)d_in[31];
  const float* ext_t_b  = (const float*)d_in[32];
  const float* ext_t_g  = (const float*)d_in[33];
  const float* ext_t_be = (const float*)d_in[34];
  const float* w1s      = (const float*)d_in[35];
  const float* w2s      = (const float*)d_in[36];
  const float* w3s      = (const float*)d_in[37];
  float* out = (float*)d_out;

  char* ws = (char*)d_ws;
  size_t off = 0;
  auto take = [&](size_t bytes) { char* p = ws + off; off += (bytes + 255) & ~(size_t)255; return p; };

  unsigned short* Xbf  = (unsigned short*)take(33554432);  // 64 x 262144 bf16
  unsigned short* Tbf  = (unsigned short*)take(33554432);
  unsigned short* pf   = (unsigned short*)take(33554432);  // reused as of_b
  unsigned short* tf   = (unsigned short*)take(33554432);
  unsigned short* pt   = (unsigned short*)take(67108864);  // reused as ot_b
  unsigned short* tt   = (unsigned short*)take(67108864);
  float* yf            = (float*)take(67108864);
  float* yt            = (float*)take(67108864);
  float* Hs            = (float*)take(8388608);
  unsigned short* Hsbf = (unsigned short*)take(4194304);
  unsigned short* psb  = (unsigned short*)take(4194304);
  unsigned short* vvb  = (unsigned short*)take(4194304);
  float* Al            = (float*)take(4194304);
  unsigned short* Aso  = (unsigned short*)take(2097152);
  float* dl            = (float*)take(8388608);            // delta_mat 2048x1024
  float* av            = (float*)take(8388608);            // AVs_mat   2048x1024
  unsigned short* rp   = (unsigned short*)take(4194304);
  unsigned short* wso  = (unsigned short*)take(4194304);
  float* osp           = (float*)take(8388608);            // 64 x 32768
  float* gf            = (float*)take(16384);
  float* ofm           = (float*)take(16384);
  float* gt            = (float*)take(65536);
  float* otm           = (float*)take(65536);
  float* stats         = (float*)take(1024);
  unsigned short* ofb  = pf;   // buffer reuse
  unsigned short* otb  = pt;

  const int NF = 262144;

  // -- prep: bf16 activation copies + zero atomic targets --
  k_cvt   <<<65536, 256, 0, stream>>>(x, Xbf, 16777216L);
  k_transp<<<65536, 256, 0, stream>>>(x, Tbf);
  k_zero  <<<64,   256, 0, stream>>>(gf, 4096);
  k_zero  <<<64,   256, 0, stream>>>(gt, 16384);
  k_zero  <<<4096, 256, 0, stream>>>(Al, 1048576);

  // -- feature / temporal projections (WMMA NN GEMMs) --
  k_gemm_nn<<<dim3(2048, 1), 256, 0, stream>>>(phi_f_w, Xbf, phi_f_b, pf, 64, 64, NF, 1);
  k_gemm_nn<<<dim3(2048, 1), 256, 0, stream>>>(th_f_w,  Xbf, th_f_b,  tf, 64, 64, NF, 1);
  k_gemm_nn<<<dim3(2048, 2), 256, 0, stream>>>(phi_t_w, Tbf, phi_t_b, pt, 128, 64, NF, 1);
  k_gemm_nn<<<dim3(2048, 2), 256, 0, stream>>>(th_t_w,  Tbf, th_t_b,  tt, 128, 64, NF, 1);

  // -- wide-K grams (WMMA NT split-K + f32 atomics) --
  k_gemm_nt<<<dim3(1, 1, 64), 256, 0, stream>>>(pf, tf, gf, 64, 64, (long)NF, 4096);
  k_gemm_nt<<<dim3(2, 2, 64), 256, 0, stream>>>(pt, tt, gt, 128, 128, (long)NF, 4096);

  // -- GCNs --
  k_gcn<<<1, 256, 64 * 64 * 4,   stream>>>(gf, gcnf_w1, gcnf_b1, gcnf_w2, ofm, 64);
  k_gcn<<<1, 256, 128 * 128 * 4, stream>>>(gt, gcnt_w1, gcnt_b1, gcnt_w2, otm, 128);

  // -- project back + extend convs (BN stats later) --
  k_gemm_nn<<<dim3(2048, 1), 256, 0, stream>>>(ofm, tf, nullptr, ofb, 64, 64, NF, 1);
  k_gemm_nn<<<dim3(2048, 1), 256, 0, stream>>>(ext_f_w, ofb, ext_f_b, yf, 64, 64, NF, 0);
  k_gemm_nn<<<dim3(2048, 2), 256, 0, stream>>>(otm, tt, nullptr, otb, 128, 128, NF, 1);
  k_gemm_nn<<<dim3(2048, 1), 256, 0, stream>>>(ext_t_w, otb, ext_t_b, yt, 64, 128, NF, 0);

  // -- spatial branch --
  k_dwconv<<<8192, 256, 0, stream>>>(x, sproj_w, sproj_b, Hs, Hsbf);
  k_ps    <<<8192, 256, 0, stream>>>(Hs, phi_s_w, phi_s_b, psb);
  k_vv    <<<8192, 256, 0, stream>>>(Hs, v_w, v_b, vvb);
  k_gemm_nt<<<dim3(16, 16, 1), 256, 0, stream>>>(vvb, psb, Al, 1024, 1024, 2048L, 2048);
  k_softmax<<<1024, 256, 0, stream>>>(Al, Aso);
  k_gemm_nn<<<dim3(256, 1), 256, 0, stream>>>(delta_w, Hsbf, delta_b, dl, 64, 64, 32768, 0);
  k_gemm_nn<<<dim3(8, 32),  256, 0, stream>>>(dl, Aso, nullptr, av, 2048, 1024, 1024, 0);
  k_repack<<<8192, 256, 0, stream>>>(av, rp);
  k_gemm_nn<<<dim3(256, 1), 256, 0, stream>>>(Ws_w, rp,  Ws_b, wso, 64, 64, 32768, 1);
  k_gemm_nn<<<dim3(256, 1), 256, 0, stream>>>(xi_w, wso, xi_b, osp, 64, 64, 32768, 0);

  // -- batchnorm stats + fused combine (up2 folded into gather) --
  k_rowstats<<<128, 256, 0, stream>>>(yf, yt, stats);
  k_final<<<65536, 256, 0, stream>>>(x, yf, yt, osp, stats,
                                     ext_f_g, ext_f_be, ext_t_g, ext_t_be,
                                     w1s, w2s, w3s, out);
}
